// MHA_85272280695594
// MI455X (gfx1250) — matrix operand order repaired and checked by reading, not compiled, not run on hardware
//
#include <hip/hip_runtime.h>
#include <hip/hip_bf16.h>
#include <stdint.h>

// ---- problem constants (from reference) ----
#define BATCH   2
#define SEQ     2048
#define NHEAD   16
#define HD      64
#define EMB     1024            // NHEAD*HD
#define MROWS   (BATCH*SEQ)     // 4096

// ---- CDNA5 WMMA types ----
typedef __attribute__((ext_vector_type(16))) __bf16 bf16x16;
typedef __attribute__((ext_vector_type(8)))  float  f32x8;

#if __has_builtin(__builtin_amdgcn_global_load_async_to_lds_b128)
#define ASYNC_LDS 1
#else
#define ASYNC_LDS 0
#endif

struct U4x2 { uint4 a, b; };
union U4US { uint4 u; unsigned short s[8]; };

__device__ __forceinline__ unsigned short f32_to_bf16_rne(float f) {
  union { float f; unsigned u; } v; v.f = f;
  unsigned lsb = (v.u >> 16) & 1u;
  return (unsigned short)((v.u + 0x7FFFu + lsb) >> 16);
}

__device__ __forceinline__ bf16x16 frag_from(uint4 lo, uint4 hi) {
  U4x2 t; t.a = lo; t.b = hi;
  return __builtin_bit_cast(bf16x16, t);      // pure register re-interpretation
}

// A-fragment (16x32, row M = lane%16): K = e + 16*hf
__device__ __forceinline__ bf16x16 load_a_frag(const unsigned short* p, int hf) {
  uint4 lo = *(const uint4*)(p + 16 * hf);
  uint4 hi = *(const uint4*)(p + 16 * hf + 8);
  return frag_from(lo, hi);
}
// B-fragment (32x16, col N = lane%16): K = e + 16*hf -> 32 contiguous bytes
__device__ __forceinline__ bf16x16 load_b_frag(const unsigned short* p, int hf) {
  return *(const bf16x16*)(p + 16 * hf);
}

__device__ __forceinline__ f32x8 wmma_bf16(bf16x16 a, bf16x16 b, f32x8 c) {
  return __builtin_amdgcn_wmma_f32_16x16x32_bf16(false, a, false, b, (short)0, c,
                                                 false, false);
}

// Stage one 32x64 bf16 K-tile (4KB, contiguous in global) into LDS.
// 128 threads x 32B each. Async path uses gfx1250 GLOBAL_LOAD_ASYNC_TO_LDS.
// Builtin (diag-confirmed): param is pointer to native clang vector
// 'int __attribute__((vector_size(16)))' in AS1 (global) / AS3 (LDS).
__device__ __forceinline__ void stage_k_tile(const unsigned short* gsrc,
                                             unsigned short* ldst, int tid) {
  const unsigned short* g = gsrc + tid * 16;
  unsigned short* l = ldst + tid * 16;
#if ASYNC_LDS
  typedef int vint4 __attribute__((vector_size(16)));
  typedef __attribute__((address_space(1))) vint4* gas1_v4;
  typedef __attribute__((address_space(3))) vint4* las3_v4;
  __builtin_amdgcn_global_load_async_to_lds_b128((gas1_v4)g, (las3_v4)l, 0, 0);
  __builtin_amdgcn_global_load_async_to_lds_b128((gas1_v4)(g + 8), (las3_v4)(l + 8),
                                                 0, 0);
#else
  uint4 v0 = *(const uint4*)g;
  uint4 v1 = *(const uint4*)(g + 8);
  *(uint4*)l = v0;
  *(uint4*)(l + 8) = v1;
#endif
}

// ---- kernel 1: f32 -> bf16 elementwise ----
__global__ void mha_cvt_bf16(const float* __restrict__ in,
                             unsigned short* __restrict__ out, int n) {
  int i = blockIdx.x * blockDim.x + threadIdx.x;
  int st = gridDim.x * blockDim.x;
  for (; i < n; i += st) out[i] = f32_to_bf16_rne(in[i]);
}

// ---- kernel 2: GEMM  out[M,N] = (A[M,K] * W[N,K]^T + bias) * scale ----
// 64x64 output tile per wave: 4x4 C-tiles, 16 WMMAs per 32-K step (4 A-frags,
// 4 B-frags reused 4x each -> matrix-pipe bound, per the roofline).
// MODE 1: store bf16 head-split [B, NHEAD, SEQ, HD]
// MODE 2: store f32 row-major [M, N]
template <int MODE>
__global__ __launch_bounds__(256) void mha_gemm_wmma(
    const unsigned short* __restrict__ A,
    const unsigned short* __restrict__ Bw,   // [N,K] row-major
    const float* __restrict__ bias,
    void* __restrict__ outp, int M, int N, int K, float outScale) {
  const int lane = threadIdx.x & 31;
  const int wv   = threadIdx.x >> 5;
  const int hf   = lane >> 4;
  const int ln16 = lane & 15;

  const int tilesN = N >> 6;                    // 64-wide tiles
  const int tile = blockIdx.x * 8 + wv;         // 8 waves/WG, one 64x64 tile each
  const int tm = tile / tilesN, tn = tile % tilesN;

  const unsigned short* abase = A  + (size_t)(tm * 64 + ln16) * K;
  const unsigned short* bbase = Bw + (size_t)(tn * 64 + ln16) * K;

  f32x8 acc[4][4] = {};
  for (int kk = 0; kk < K; kk += 32) {
    if (kk + 256 < K) {
      __builtin_prefetch(abase + kk + 256, 0, 3);
      __builtin_prefetch(bbase + kk + 256, 0, 3);
    }
    bf16x16 aF[4], bF[4];
#pragma unroll
    for (int i = 0; i < 4; ++i)
      aF[i] = load_a_frag(abase + (size_t)i * 16 * K + kk, hf);
#pragma unroll
    for (int j = 0; j < 4; ++j)
      bF[j] = load_b_frag(bbase + (size_t)j * 16 * K + kk, hf);
#pragma unroll
    for (int i = 0; i < 4; ++i)
#pragma unroll
      for (int j = 0; j < 4; ++j)
        acc[i][j] = wmma_bf16(aF[i], bF[j], acc[i][j]);
  }

#pragma unroll
  for (int j = 0; j < 4; ++j) {
    const int col = tn * 64 + j * 16 + ln16;
    const float bv = bias[col];
#pragma unroll
    for (int i = 0; i < 4; ++i) {
#pragma unroll
      for (int r = 0; r < 8; ++r) {
        const int row = tm * 64 + i * 16 + r + 8 * hf;   // C layout: M=r+8*hf
        const float val = (acc[i][j][r] + bv) * outScale;
        if (MODE == 1) {
          unsigned short* o = (unsigned short*)outp;
          int b = row / SEQ, s = row - b * SEQ;
          int h = col >> 6, d = col & 63;
          o[(((size_t)(b * NHEAD + h)) * SEQ + s) * HD + d] = f32_to_bf16_rne(val);
        } else {
          float* o = (float*)outp;
          o[(size_t)row * N + col] = val;
        }
      }
    }
  }
}

// ---- kernel 3: causal flash attention, one (b,h, 64-query) tile per WG ----
// Q pre-scaled by 1/sqrt(HD) in its projection. K tiles double-buffered in LDS
// via async loads; V tiles transposed into LDS for contiguous B-fragments.
__global__ __launch_bounds__(128) void mha_flash_wmma(
    const unsigned short* __restrict__ Q,   // [B*H, S, HD] bf16 (pre-scaled)
    const unsigned short* __restrict__ Kt,  // [B*H, S, HD] bf16
    const unsigned short* __restrict__ V,   // [B*H, S, HD] bf16
    unsigned short* __restrict__ ctx) {     // [B, S, EMB]  bf16 (merged heads)
  __shared__ unsigned short Kl[2][32 * HD];    // double-buffered K tile [n][d]
  __shared__ unsigned short Vt[HD * 32];       // V tile transposed: [d][n]
  __shared__ unsigned short Pl[4][16 * 32];    // per-wave P tile (m x n)

  const int tid  = threadIdx.x;
  const int lane = tid & 31, wv = tid >> 5;
  const int hf = lane >> 4, ln16 = lane & 15;

  const int bh   = blockIdx.x >> 5;            // 32 q-tiles per (b,h)
  const int qblk = blockIdx.x & 31;
  const int q0   = qblk * 64;
  const int q0w  = q0 + wv * 16;

  const unsigned short* Qh = Q  + (size_t)bh * SEQ * HD;
  const unsigned short* Kh = Kt + (size_t)bh * SEQ * HD;
  const unsigned short* Vh = V  + (size_t)bh * SEQ * HD;

  // Q tile (16 rows x 64 d) as two A-fragments, resident for the whole pass
  bf16x16 qA0 = load_a_frag(Qh + (size_t)(q0w + ln16) * HD + 0,  hf);
  bf16x16 qA1 = load_a_frag(Qh + (size_t)(q0w + ln16) * HD + 32, hf);

  float m_i[8], l_i[8];
  f32x8 acc[4] = {};
#pragma unroll
  for (int r = 0; r < 8; ++r) { m_i[r] = -1e30f; l_i[r] = 0.f; }

  const int nblk = (q0 >> 5) + 2;              // causal: key blocks up to diagonal
  stage_k_tile(Kh, &Kl[0][0], tid);            // prefetch K tile for block 0

  for (int blk = 0; blk < nblk; ++blk) {
    const int kb = blk * 32;
    const int cur = blk & 1;

#if ASYNC_LDS
    asm volatile("s_wait_asynccnt 0x0" ::: "memory");  // this block's K landed
#endif
    __syncthreads();                           // K visible; prev readers done

    {
      // cooperative transpose-load of V tile: 32 keys x 64 d -> Vt[d][n]
      const int n  = tid >> 2;
      const int db = (tid & 3) * 16;
      const unsigned short* vp = Vh + (size_t)(kb + n) * HD + db;
      U4US v0, v1;
      v0.u = *(const uint4*)vp;
      v1.u = *(const uint4*)(vp + 8);
#pragma unroll
      for (int i = 0; i < 8; ++i) {
        Vt[(db + i) * 32 + n]     = v0.s[i];
        Vt[(db + 8 + i) * 32 + n] = v1.s[i];
      }
    }
    if (blk + 1 < nblk)                        // pipeline: prefetch next K tile
      stage_k_tile(Kh + (size_t)(kb + 32) * HD, &Kl[cur ^ 1][0], tid);
    __syncthreads();                           // Vt ready

    // S = Q * K^T : two 16-key score tiles, contraction over d=64
    f32x8 st[2];
#pragma unroll
    for (int t = 0; t < 2; ++t) {
      f32x8 c = {};
      const unsigned short* krow = &Kl[cur][(t * 16 + ln16) * HD];
      c = wmma_bf16(qA0, load_b_frag(krow,      hf), c);
      c = wmma_bf16(qA1, load_b_frag(krow + 32, hf), c);
      st[t] = c;
    }

    // online softmax (rows live in lanes 0-15 / 16-31 of each accumulator VGPR)
#pragma unroll
    for (int r = 0; r < 8; ++r) {
      const int row = q0w + r + 8 * hf;
      float s0 = st[0][r];
      float s1 = st[1][r];
      if (kb + ln16      > row) s0 = -1e30f;   // causal mask
      if (kb + 16 + ln16 > row) s1 = -1e30f;
      float bm = fmaxf(s0, s1);
#pragma unroll
      for (int m = 1; m <= 8; m <<= 1) bm = fmaxf(bm, __shfl_xor(bm, m, 32));
      const float mn    = fmaxf(m_i[r], bm);
      const float alpha = __expf(m_i[r] - mn);
      const float p0 = __expf(s0 - mn);
      const float p1 = __expf(s1 - mn);
      float rs = p0 + p1;
#pragma unroll
      for (int m = 1; m <= 8; m <<= 1) rs += __shfl_xor(rs, m, 32);
      l_i[r] = l_i[r] * alpha + rs;
      m_i[r] = mn;
#pragma unroll
      for (int t = 0; t < 4; ++t) acc[t][r] *= alpha;
      const int mrow = r + 8 * hf;
      Pl[wv][mrow * 32 + ln16]      = f32_to_bf16_rne(p0);
      Pl[wv][mrow * 32 + 16 + ln16] = f32_to_bf16_rne(p1);
    }

    // wave-private LDS RAW: make sure P stores completed before fragment reads
    asm volatile("s_wait_dscnt 0" ::: "memory");

    // O += P(16x32) * V(32x64): 4 WMMAs, B-fragments from transposed LDS tile
    bf16x16 pA = load_a_frag(&Pl[wv][ln16 * 32], hf);
#pragma unroll
    for (int t = 0; t < 4; ++t) {
      bf16x16 vB = load_b_frag(&Vt[(t * 16 + ln16) * 32], hf);
      acc[t] = wmma_bf16(pA, vB, acc[t]);
    }
  }

  // epilogue: normalize and store merged-head bf16 ctx [B, S, EMB]
  const int b = bh >> 4, hh = bh & 15;
#pragma unroll
  for (int r = 0; r < 8; ++r) {
    const float inv = 1.0f / l_i[r];
    const int row = q0w + r + 8 * hf;
#pragma unroll
    for (int t = 0; t < 4; ++t) {
      const int col = hh * HD + t * 16 + ln16;
      ctx[((size_t)(b * SEQ + row)) * EMB + col] = f32_to_bf16_rne(acc[t][r] * inv);
    }
  }
}

// ---- host-side orchestration ----
extern "C" void kernel_launch(void* const* d_in, const int* in_sizes, int n_in,
                              void* d_out, int out_size, void* d_ws, size_t ws_size,
                              hipStream_t stream) {
  (void)in_sizes; (void)n_in; (void)out_size; (void)ws_size;
  const float* x  = (const float*)d_in[0];
  const float* Wq = (const float*)d_in[1];
  const float* bq = (const float*)d_in[2];
  const float* Wk = (const float*)d_in[3];
  const float* bk = (const float*)d_in[4];
  const float* Wv = (const float*)d_in[5];
  const float* bv = (const float*)d_in[6];
  const float* Wo = (const float*)d_in[7];
  const float* bo = (const float*)d_in[8];

  char* ws = (char*)d_ws;
  size_t off = 0;
  auto alloc_us = [&](size_t elems) -> unsigned short* {
    unsigned short* p = (unsigned short*)(ws + off);
    off += elems * sizeof(unsigned short);
    return p;
  };
  unsigned short* xb   = alloc_us((size_t)MROWS * EMB);
  unsigned short* wqb  = alloc_us((size_t)EMB * EMB);
  unsigned short* wkb  = alloc_us((size_t)EMB * EMB);
  unsigned short* wvb  = alloc_us((size_t)EMB * EMB);
  unsigned short* wob  = alloc_us((size_t)EMB * EMB);
  unsigned short* qb   = alloc_us((size_t)MROWS * EMB);
  unsigned short* kbuf = alloc_us((size_t)MROWS * EMB);
  unsigned short* vbuf = alloc_us((size_t)MROWS * EMB);
  unsigned short* ctxb = alloc_us((size_t)MROWS * EMB);

  mha_cvt_bf16<<<2048, 256, 0, stream>>>(x,  xb,  MROWS * EMB);
  mha_cvt_bf16<<<1024, 256, 0, stream>>>(Wq, wqb, EMB * EMB);
  mha_cvt_bf16<<<1024, 256, 0, stream>>>(Wk, wkb, EMB * EMB);
  mha_cvt_bf16<<<1024, 256, 0, stream>>>(Wv, wvb, EMB * EMB);
  mha_cvt_bf16<<<1024, 256, 0, stream>>>(Wo, wob, EMB * EMB);

  const int gemmBlocks = (MROWS / 64) * (EMB / 64) / 8;   // 128 WGs, 8 waves each
  // Q projection carries the 1/sqrt(HD) softmax scale
  mha_gemm_wmma<1><<<gemmBlocks, 256, 0, stream>>>(xb, wqb, bq, qb,   MROWS, EMB, EMB, 0.125f);
  mha_gemm_wmma<1><<<gemmBlocks, 256, 0, stream>>>(xb, wkb, bk, kbuf, MROWS, EMB, EMB, 1.0f);
  mha_gemm_wmma<1><<<gemmBlocks, 256, 0, stream>>>(xb, wvb, bv, vbuf, MROWS, EMB, EMB, 1.0f);

  mha_flash_wmma<<<BATCH * NHEAD * (SEQ / 64), 128, 0, stream>>>(qb, kbuf, vbuf, ctxb);

  mha_gemm_wmma<2><<<gemmBlocks, 256, 0, stream>>>(ctxb, wob, bo, d_out, MROWS, EMB, EMB, 1.0f);
}
